// MoE_7146825580883
// MI455X (gfx1250) — compile-verified, matrix operand order checked by reading, not run on hardware
//
#include <hip/hip_runtime.h>

// Problem constants (match reference)
#define L_TOK 4096
#define H_DIM 1024
#define F_DIM 512
#define E_NUM 8

typedef __attribute__((ext_vector_type(8)))  __bf16 bf16x8;
typedef __attribute__((ext_vector_type(16))) __bf16 bf16x16;
typedef __attribute__((ext_vector_type(8)))  float  f32x8;

// ---- fp32 -> bf16 with round-to-nearest-even (pure integer ops) ----
__device__ __forceinline__ unsigned short f2bf_bits(float f) {
    union { float f; unsigned u; } c; c.f = f;
    unsigned u = c.u;
    unsigned r = u + 0x7FFFu + ((u >> 16) & 1u);
    return (unsigned short)(r >> 16);
}
__device__ __forceinline__ __bf16 f2bf(float f) {
    union { unsigned short s; __bf16 b; } o; o.s = f2bf_bits(f);
    return o.b;
}

// Load one 16-lane-striped B fragment (K-major row of length Kdim) from global.
__device__ __forceinline__ bf16x16 load_bfrag(const __bf16* __restrict__ base,
                                              int row, int Kdim, int ka) {
    const __bf16* wr = base + (size_t)row * Kdim + ka;
    bf16x8 blo = *(const bf16x8*)(wr);
    bf16x8 bhi = *(const bf16x8*)(wr + 16);
    return __builtin_shufflevector(blo, bhi,
        0,1,2,3,4,5,6,7,8,9,10,11,12,13,14,15);
}

// =====================================================================
// Kernel 1: convert fp32 expert weights to bf16 (one-time per call)
// =====================================================================
__global__ __launch_bounds__(256) void moe_convert_kernel(
    const float* __restrict__ w13, const float* __restrict__ w2,
    unsigned short* __restrict__ w13b, unsigned short* __restrict__ w2b)
{
    const size_t N13 = (size_t)E_NUM * 2 * F_DIM * H_DIM; // 8,388,608
    const size_t N2  = (size_t)E_NUM * H_DIM * F_DIM;     // 4,194,304
    size_t i = (size_t)blockIdx.x * 256 + threadIdx.x;
    if (i < N13) w13b[i] = f2bf_bits(w13[i]);
    if (i < N2)  w2b[i]  = f2bf_bits(w2[i]);
}

// =====================================================================
// Kernel 2: router — logits, softmax(top-2 renorm), expert buckets
//   8 threads per token (one per expert), 32 tokens per block.
// =====================================================================
__global__ __launch_bounds__(256) void moe_router_kernel(
    const float* __restrict__ x, const float* __restrict__ gate_w,
    int* __restrict__ counts, int* __restrict__ bucket,
    float* __restrict__ topk_prob)
{
    __shared__ float s_logit[32][E_NUM];
    const int tid = threadIdx.x;
    const int tok = blockIdx.x * 32 + (tid >> 3);
    const int e   = tid & 7;

    const float* xr = x + (size_t)tok * H_DIM;
    const float* gr = gate_w + (size_t)e * H_DIM;
    float acc = 0.0f;
    for (int i = 0; i < H_DIM; i += 4) {
        float4 xv = *(const float4*)(xr + i);
        float4 gv = *(const float4*)(gr + i);
        acc += xv.x * gv.x + xv.y * gv.y + xv.z * gv.z + xv.w * gv.w;
    }
    s_logit[tid >> 3][e] = acc;
    __syncthreads();

    if (tid < 32) {
        const int t = blockIdx.x * 32 + tid;
        float l[E_NUM];
        float mx = -3.0e38f;
        #pragma unroll
        for (int i = 0; i < E_NUM; ++i) { l[i] = s_logit[tid][i]; mx = fmaxf(mx, l[i]); }
        #pragma unroll
        for (int i = 0; i < E_NUM; ++i) l[i] = __expf(l[i] - mx);
        // top-2 (softmax denominator cancels under renormalization)
        int i0 = 0;
        #pragma unroll
        for (int i = 1; i < E_NUM; ++i) if (l[i] > l[i0]) i0 = i;
        int i1 = (i0 == 0) ? 1 : 0;
        #pragma unroll
        for (int i = 0; i < E_NUM; ++i) if (i != i0 && l[i] > l[i1]) i1 = i;
        float p0 = l[i0], p1 = l[i1];
        float inv = 1.0f / (p0 + p1);
        topk_prob[t * 2 + 0] = p0 * inv;
        topk_prob[t * 2 + 1] = p1 * inv;
        int pos0 = atomicAdd(&counts[i0], 1);
        bucket[i0 * L_TOK + pos0] = t * 2 + 0;   // dest slot = token*2 + k
        int pos1 = atomicAdd(&counts[i1], 1);
        bucket[i1 * L_TOK + pos1] = t * 2 + 1;
    }
}

// =====================================================================
// Kernel 3: grouped expert FFN (SwiGLU) on 16-token tiles, WMMA bf16
//   block = 256 threads = 8 waves; wave w owns 128 output columns.
//   B fragments are software-pipelined one step ahead of the WMMA
//   that consumes them, so each load has a full WMMA to complete.
// =====================================================================
__global__ __launch_bounds__(256) void moe_ffn_kernel(
    const float* __restrict__ x,
    const unsigned short* __restrict__ w13u,
    const unsigned short* __restrict__ w2u,
    const int* __restrict__ counts,
    const int* __restrict__ bucket,
    float* __restrict__ ybuf)
{
    const int e    = blockIdx.y;
    const int n_e  = counts[e];
    const int row0 = blockIdx.x * 16;
    if (row0 >= n_e) return;                       // block-uniform: EXEC stays full
    const int valid = (n_e - row0 < 16) ? (n_e - row0) : 16;

    const __bf16* __restrict__ w13b = (const __bf16*)w13u;
    const __bf16* __restrict__ w2b  = (const __bf16*)w2u;

    __shared__ int    s_dest[16];
    __shared__ __bf16 s_x[16][H_DIM + 8];          // x tile, later reused as h tile
    __shared__ __bf16 s_act[16][F_DIM + 8];        // SwiGLU activations

    const int tid = threadIdx.x;
    if (tid < 16) {
        int rr = (tid < valid) ? tid : (valid - 1);   // clamp padded rows to a real token
        s_dest[tid] = bucket[e * L_TOK + row0 + rr];
    }
    __syncthreads();

    // ---- stage x tile as bf16: 16 rows x 1024 cols (16 threads/row, 64 cols each)
    {
        const int r  = tid >> 4;
        const int c0 = (tid & 15) * 64;
        const int tok = s_dest[r] >> 1;
        const float* xr = x + (size_t)tok * H_DIM + c0;
        #pragma unroll
        for (int c = 0; c < 64; c += 4) {
            float4 v = *(const float4*)(xr + c);
            s_x[r][c0 + c + 0] = f2bf(v.x);
            s_x[r][c0 + c + 1] = f2bf(v.y);
            s_x[r][c0 + c + 2] = f2bf(v.z);
            s_x[r][c0 + c + 3] = f2bf(v.w);
        }
    }
    __syncthreads();

    const int wave = tid >> 5;
    const int lane = tid & 31;
    const int lrow = lane & 15;          // M for A-frag, N for B-frag
    const int koff = (lane >> 4) * 8;    // K sub-slice per lane half
    const int nb0  = wave * 128;         // this wave's 128 output columns

    // ================= Phase A: h[16 x 1024] = x_tile @ w13_e^T =================
    f32x8 acc[8] = {};
    const __bf16* wA = w13b + (size_t)e * (2 * F_DIM) * H_DIM;
    {
        bf16x16 bcur = load_bfrag(wA, nb0 + lrow, H_DIM, koff);
        for (int k0 = 0; k0 < H_DIM; k0 += 32) {
            const int ka = k0 + koff;
            const int kn = (k0 + 32 < H_DIM) ? (ka + 32) : koff;  // uniform wrap
            __builtin_prefetch(wA + (size_t)(nb0 + lrow) * H_DIM + kn + 32, 0, 1);
            bf16x8 alo = *(const bf16x8*)&s_x[lrow][ka];
            bf16x8 ahi = *(const bf16x8*)&s_x[lrow][ka + 16];
            bf16x16 afrag = __builtin_shufflevector(alo, ahi,
                0,1,2,3,4,5,6,7,8,9,10,11,12,13,14,15);
            #pragma unroll
            for (int t = 0; t < 8; ++t) {
                bf16x16 bnext = (t < 7)
                    ? load_bfrag(wA, nb0 + (t + 1) * 16 + lrow, H_DIM, ka)
                    : load_bfrag(wA, nb0 + lrow, H_DIM, kn);
                acc[t] = __builtin_amdgcn_wmma_f32_16x16x32_bf16(
                    false, afrag, false, bcur, (short)0, acc[t], false, false);
                bcur = bnext;
            }
        }
    }
    __syncthreads();   // everyone done reading s_x -> safe to reuse as h tile

    // write h (bf16) back into s_x (D layout: vgpr r -> M=r (lanes 0-15) / M=8+r)
    {
        const int mbase = (lane >> 4) * 8;
        #pragma unroll
        for (int t = 0; t < 8; ++t) {
            const int n = nb0 + t * 16 + lrow;
            #pragma unroll
            for (int r = 0; r < 8; ++r)
                s_x[mbase + r][n] = f2bf(acc[t][r]);
        }
    }
    __syncthreads();

    // ---- SwiGLU: act[m][f] = silu(h[m][f]) * h[m][F+f]
    {
        const int r  = tid >> 4;
        const int c0 = (tid & 15) * 32;
        #pragma unroll 4
        for (int c = 0; c < 32; ++c) {
            float g = (float)s_x[r][c0 + c];
            float u = (float)s_x[r][F_DIM + c0 + c];
            float sg = g / (1.0f + __expf(-g));
            s_act[r][c0 + c] = f2bf(sg * u);
        }
    }
    __syncthreads();

    // ================= Phase B: y[16 x 1024] = act @ w2_e^T =================
    f32x8 acc2[8] = {};
    const __bf16* wB = w2b + (size_t)e * H_DIM * F_DIM;
    {
        bf16x16 bcur = load_bfrag(wB, nb0 + lrow, F_DIM, koff);
        for (int k0 = 0; k0 < F_DIM; k0 += 32) {
            const int ka = k0 + koff;
            const int kn = (k0 + 32 < F_DIM) ? (ka + 32) : koff;  // uniform wrap
            __builtin_prefetch(wB + (size_t)(nb0 + lrow) * F_DIM + kn + 32, 0, 1);
            bf16x8 alo = *(const bf16x8*)&s_act[lrow][ka];
            bf16x8 ahi = *(const bf16x8*)&s_act[lrow][ka + 16];
            bf16x16 afrag = __builtin_shufflevector(alo, ahi,
                0,1,2,3,4,5,6,7,8,9,10,11,12,13,14,15);
            #pragma unroll
            for (int t = 0; t < 8; ++t) {
                bf16x16 bnext = (t < 7)
                    ? load_bfrag(wB, nb0 + (t + 1) * 16 + lrow, F_DIM, ka)
                    : load_bfrag(wB, nb0 + lrow, F_DIM, kn);
                acc2[t] = __builtin_amdgcn_wmma_f32_16x16x32_bf16(
                    false, afrag, false, bcur, (short)0, acc2[t], false, false);
                bcur = bnext;
            }
        }
    }

    // ---- scatter fp32 results to unique (token,k) rows of ybuf
    {
        const int mbase = (lane >> 4) * 8;
        #pragma unroll
        for (int t = 0; t < 8; ++t) {
            const int n = nb0 + t * 16 + lrow;
            #pragma unroll
            for (int r = 0; r < 8; ++r) {
                const int m = mbase + r;
                if (m < valid)
                    ybuf[(size_t)s_dest[m] * H_DIM + n] = acc2[t][r];
            }
        }
    }
}

// =====================================================================
// Kernel 4: combine — out[l] = p0*y[2l] + p1*y[2l+1]
// =====================================================================
__global__ __launch_bounds__(256) void moe_combine_kernel(
    const float* __restrict__ ybuf, const float* __restrict__ topk_prob,
    float* __restrict__ out)
{
    const size_t idx = (size_t)blockIdx.x * 256 + threadIdx.x;  // over L*H
    const int l = (int)(idx >> 10);
    const int h = (int)(idx & (H_DIM - 1));
    float p0 = topk_prob[2 * l + 0];
    float p1 = topk_prob[2 * l + 1];
    out[idx] = p0 * ybuf[(size_t)(2 * l + 0) * H_DIM + h]
             + p1 * ybuf[(size_t)(2 * l + 1) * H_DIM + h];
}

// =====================================================================
extern "C" void kernel_launch(void* const* d_in, const int* in_sizes, int n_in,
                              void* d_out, int out_size, void* d_ws, size_t ws_size,
                              hipStream_t stream) {
    const float* x      = (const float*)d_in[0];   // [L,H]
    const float* gate_w = (const float*)d_in[1];   // [E,H]
    const float* w13    = (const float*)d_in[2];   // [E,2F,H]
    const float* w2     = (const float*)d_in[3];   // [E,H,F]
    float* out = (float*)d_out;

    // workspace layout
    char* ws = (char*)d_ws;
    size_t o = 0;
    unsigned short* w13b = (unsigned short*)(ws + o); o += (size_t)E_NUM * 2 * F_DIM * H_DIM * 2;
    unsigned short* w2b  = (unsigned short*)(ws + o); o += (size_t)E_NUM * H_DIM * F_DIM * 2;
    float* ybuf   = (float*)(ws + o); o += (size_t)2 * L_TOK * H_DIM * 4;
    int*   bucket = (int*)(ws + o);   o += (size_t)E_NUM * L_TOK * 4;
    float* tkp    = (float*)(ws + o); o += (size_t)L_TOK * 2 * 4;
    int*   counts = (int*)(ws + o);   o += 128;

    hipMemsetAsync(counts, 0, E_NUM * sizeof(int), stream);

    // 1) weights -> bf16
    moe_convert_kernel<<<32768, 256, 0, stream>>>(w13, w2, w13b, w2b);
    // 2) router + buckets
    moe_router_kernel<<<L_TOK / 32, 256, 0, stream>>>(x, gate_w, counts, bucket, tkp);
    // 3) grouped FFN (tiles beyond each expert's count exit immediately)
    moe_ffn_kernel<<<dim3(L_TOK / 16, E_NUM), 256, 0, stream>>>(x, w13b, w2b, counts, bucket, ybuf);
    // 4) weighted combine
    moe_combine_kernel<<<(L_TOK * H_DIM) / 256, 256, 0, stream>>>(ybuf, tkp, out);
}
